// transformer_block_61710090109659
// MI455X (gfx1250) — matrix operand order repaired and checked by reading, not compiled, not run on hardware
//
#include <hip/hip_runtime.h>
#include <stdint.h>
#include <stddef.h>

// ---------------------------------------------------------------------------
// Types & helpers
// ---------------------------------------------------------------------------
typedef __attribute__((ext_vector_type(16))) __bf16 v16bf;
typedef __attribute__((ext_vector_type(8)))  float  v8f;

union Frag16 { uint4 q[2]; v16bf v; };

__device__ __forceinline__ uint32_t bfbits(float f) {   // rounded bf16 in [31:16]
  union { float f; uint32_t u; } c; c.f = f;
  return c.u + 0x7FFFu + ((c.u >> 16) & 1u);
}
__device__ __forceinline__ __bf16 f2bf(float f) {
  uint16_t h = (uint16_t)(bfbits(f) >> 16);
  __bf16 b; __builtin_memcpy(&b, &h, 2); return b;
}
// pack two floats to packed bf16 pair: round-bias add + v_perm_b32 byte blend
__device__ __forceinline__ uint32_t pk2bf(float lo, float hi) {
  union { float f; uint32_t u; } a, b; a.f = lo; b.f = hi;
  return __builtin_amdgcn_perm(b.u + 0x8000u, a.u + 0x8000u, 0x07060302u);
}
__device__ __forceinline__ float bflo(uint32_t p) {
  union { uint32_t u; float f; } c; c.u = p << 16; return c.f;
}
__device__ __forceinline__ float bfhi(uint32_t p) {
  union { uint32_t u; float f; } c; c.u = p & 0xFFFF0000u; return c.f;
}

// 16 contiguous bf16 -> fragment (2 x b128)
__device__ __forceinline__ v16bf ld_frag16(const __bf16* p) {
  Frag16 f;
  f.q[0] = *(const uint4*)p;
  f.q[1] = *(const uint4*)(p + 8);
  return f.v;
}
// A-style fragment: 16 contiguous elements at p+b0
__device__ __forceinline__ v16bf ld_afrag(const __bf16* p, int b0) {
  Frag16 f;
  f.q[0] = *(const uint4*)(p + b0);
  f.q[1] = *(const uint4*)(p + b0 + 8);
  return f.v;
}

__device__ __forceinline__ v8f wmma_bf16(v16bf a, v16bf b, v8f c) {
  return __builtin_amdgcn_wmma_f32_16x16x32_bf16(false, a, false, b, (short)0, c, false, false);
}

// CDNA5 async global->LDS copy (16B per lane), tracked by ASYNCcnt
__device__ __forceinline__ void async_ld_b128(uint32_t lds_byte_addr, const void* gaddr) {
  asm volatile("global_load_async_to_lds_b128 %0, %1, off"
               :: "v"(lds_byte_addr), "v"(gaddr) : "memory");
}
__device__ __forceinline__ void wait_async0() {
  asm volatile("s_wait_asynccnt 0" ::: "memory");
}

// Problem constants
#define CB 4
#define CS 2048
#define CD 1024
#define CF 4096
#define CH 16

// ---------------------------------------------------------------------------
// RMSNorm (fp32 in, bf16 out): one block (256 thr) per row of D=1024
// ---------------------------------------------------------------------------
__global__ __launch_bounds__(256)
void rmsnorm_kernel(const float* __restrict__ x, const float* __restrict__ gamma,
                    __bf16* __restrict__ out, int D) {
  const int row = blockIdx.x;
  const float* xr = x + (size_t)row * D;
  __bf16* orow = out + (size_t)row * D;

  const int i4 = threadIdx.x * 4;
  float4 xv = *(const float4*)(xr + i4);
  float s = xv.x * xv.x + xv.y * xv.y + xv.z * xv.z + xv.w * xv.w;
  #pragma unroll
  for (int off = 16; off > 0; off >>= 1) s += __shfl_xor(s, off, 32);

  __shared__ float wsum[8];
  const int lane = threadIdx.x & 31, wv = threadIdx.x >> 5;
  if (lane == 0) wsum[wv] = s;
  __syncthreads();
  float tot = 0.f;
  #pragma unroll
  for (int i = 0; i < 8; ++i) tot += wsum[i];
  const float rn = rsqrtf(tot / (float)D + 1e-5f);

  float4 gv = *(const float4*)(gamma + i4);
  uint2 o;
  o.x = pk2bf(xv.x * rn * gv.x, xv.y * rn * gv.y);
  o.y = pk2bf(xv.z * rn * gv.z, xv.w * rn * gv.w);
  *(uint2*)(orow + i4) = o;
}

// ---------------------------------------------------------------------------
// Tiled WMMA GEMM: C[M,N] = A[M,K](bf16 row-major) * W[N,K]^T (fp32 row-major)
//  block = 256 thr (8 waves), tile 128x128, K-step 32, double-buffered B
//  wave (mg = wv>>1, ng = wv&1) computes 32 rows x 64 cols:
//    2 A-fragments x 4 B-fragments -> 8 WMMA, each B-frag reused twice
//  STORE_MODE: 0 = f32 to Cf, 1 = bf16 to Cb, 2 = bf16 transposed-V layout
// ---------------------------------------------------------------------------
template<int STORE_MODE, bool ADD_RES>
__global__ __launch_bounds__(256)
void gemm_bf16_wmma(const __bf16* __restrict__ A, const float* __restrict__ W,
                    float* __restrict__ Cf, __bf16* __restrict__ Cb,
                    const float* __restrict__ Res, int M, int N, int K) {
  // B tile in FRAGMENT layout: [(nsg*32 + lane)*16 + e], nsg = 0..7, dbl-buf
  __shared__ alignas(16) __bf16 Bsf[2][8 * 32 * 16];

  const int tid = threadIdx.x, lane = tid & 31, wv = tid >> 5;
  const int mg = wv >> 1, ng = wv & 1;
  const int n0 = blockIdx.x * 128, m0 = blockIdx.y * 128;

  v8f acc[2][4] = {};

  // B staging: thread t covers (n = t>>1, k-half = t&1) -> one lane's 16 elems
  const int bn = tid >> 1;                 // 0..127
  const int bhalf = tid & 1;
  const int bns = bn >> 4;                 // 0..7
  const int blane = (bn & 15) + bhalf * 16;
  const float* wrow = W + (size_t)(n0 + bn) * K + bhalf * 16;
  __bf16* bdst[2] = { &Bsf[0][(bns * 32 + blane) * 16],
                      &Bsf[1][(bns * 32 + blane) * 16] };

  // A fragment sources: two row sets per wave
  const __bf16* arow0 = A + (size_t)(m0 + mg * 32 + (lane & 15)) * K;
  const __bf16* arow1 = arow0 + (size_t)16 * K;
  const int ab0 = (lane < 16) ? 0 : 16;

  auto stageB = [&](int kk, __bf16* dst) {
    const float4* wg = (const float4*)(wrow + kk);
    float4 w0 = wg[0], w1 = wg[1], w2 = wg[2], w3 = wg[3];
    uint4 d0, d1;
    d0.x = pk2bf(w0.x, w0.y); d0.y = pk2bf(w0.z, w0.w);
    d0.z = pk2bf(w1.x, w1.y); d0.w = pk2bf(w1.z, w1.w);
    d1.x = pk2bf(w2.x, w2.y); d1.y = pk2bf(w2.z, w2.w);
    d1.z = pk2bf(w3.x, w3.y); d1.w = pk2bf(w3.z, w3.w);
    *(uint4*)dst = d0;
    *(uint4*)(dst + 8) = d1;
  };

  const int kTiles = K / 32;
  stageB(0, bdst[0]);
  __syncthreads();

  for (int kt = 0; kt < kTiles; ++kt) {
    const int kk = kt * 32;
    if (kt + 1 < kTiles) {
      stageB(kk + 32, bdst[(kt & 1) ^ 1]);
      __builtin_prefetch(arow0 + kk + 64, 0, 1);           // global_prefetch_b8
      __builtin_prefetch(wrow + kk + 64, 0, 1);
    }
    const __bf16* bufc = Bsf[kt & 1];
    v16bf a0 = ld_afrag(arow0 + kk, ab0);
    v16bf a1 = ld_afrag(arow1 + kk, ab0);
    #pragma unroll
    for (int ns = 0; ns < 4; ++ns) {
      v16bf bf = ld_frag16(bufc + ((ng * 4 + ns) * 32 + lane) * 16);
      acc[0][ns] = wmma_bf16(a0, bf, acc[0][ns]);
      acc[1][ns] = wmma_bf16(a1, bf, acc[1][ns]);
    }
    __syncthreads();
  }

  // Epilogue. C/D layout: M = r + (lane<16?0:8), N = lane&15 per 16x16 tile.
  #pragma unroll
  for (int a = 0; a < 2; ++a) {
    const int row0 = m0 + mg * 32 + a * 16 + ((lane < 16) ? 0 : 8);
    #pragma unroll
    for (int ns = 0; ns < 4; ++ns) {
      const int col = n0 + ng * 64 + ns * 16 + (lane & 15);
      if (STORE_MODE == 2) {
        // V^T layout [B][H][dh][S]: 8 consecutive rows = consecutive s -> b128
        const int b = row0 >> 11, s = row0 & (CS - 1);
        const int h = col >> 6, dl = col & 63;
        __bf16* dst = Cb + ((((size_t)b * CH + h) * 64 + dl) * CS + s);
        uint4 o;
        o.x = pk2bf(acc[a][ns][0], acc[a][ns][1]);
        o.y = pk2bf(acc[a][ns][2], acc[a][ns][3]);
        o.z = pk2bf(acc[a][ns][4], acc[a][ns][5]);
        o.w = pk2bf(acc[a][ns][6], acc[a][ns][7]);
        *(uint4*)dst = o;
      } else {
        #pragma unroll
        for (int r = 0; r < 8; ++r) {
          const int row = row0 + r;
          float v = acc[a][ns][r];
          if (ADD_RES) v += Res[(size_t)row * N + col];
          if (STORE_MODE == 1) Cb[(size_t)row * N + col] = f2bf(v);
          else                 Cf[(size_t)row * N + col] = v;
        }
      }
    }
  }
}

// ---------------------------------------------------------------------------
// Causal flash attention, head dim 64. grid=(S/64, H, B), block=128 (4 waves).
//   Q, K row-major [token][D]; V transposed [B][H][64][S].
//   K/V tiles staged to LDS with GLOBAL_LOAD_ASYNC_TO_LDS_B128 (ASYNCcnt).
// ---------------------------------------------------------------------------
__global__ __launch_bounds__(128)
void attn_kernel(const __bf16* __restrict__ Qb, const __bf16* __restrict__ Kb,
                 const __bf16* __restrict__ Vt, __bf16* __restrict__ Ob,
                 int S, int D) {
  const int qt = blockIdx.x, h = blockIdx.y, b = blockIdx.z;
  const int tid = threadIdx.x, lane = tid & 31, wv = tid >> 5;
  const size_t base  = (size_t)b * S * D + (size_t)h * 64;     // Q/K/O base
  const size_t vbase = (((size_t)b * CH + h) * 64) * S;        // V^T base

  __shared__ alignas(16) __bf16 Ks[64][64];     // [key][d]
  __shared__ alignas(16) __bf16 Vs[64][64];     // [d][key]  (from V^T)
  __shared__ alignas(16) float  Ss[64][65];
  __shared__ alignas(16) __bf16 Ps[64][64];
  __shared__ alignas(16) float  mrow[64], lrow[64], alrow[64];

  // Q fragments straight from global (two b128 runs per fragment)
  const __bf16* qrow = Qb + base + (size_t)(64 * qt + wv * 16 + (lane & 15)) * D;
  const int ab0 = (lane < 16) ? 0 : 16;
  v16bf qa0 = ld_afrag(qrow, ab0);
  v16bf qa1 = ld_afrag(qrow + 32, ab0);

  if (tid < 64) { mrow[tid] = -1e30f; lrow[tid] = 0.f; }
  __syncthreads();

  v8f o[4] = {};
  const int sr = tid >> 1, sc = (tid & 1) * 32;   // staging coords
  const uint32_t ldsK = (uint32_t)(uintptr_t)&Ks[sr][sc];
  const uint32_t ldsV = (uint32_t)(uintptr_t)&Vs[sr][sc];

  for (int kt = 0; kt <= qt; ++kt) {
    // Async-stage K (row-major) and V (d-major) tiles: 4 x async b128 each
    {
      const __bf16* gk = Kb + base + (size_t)(64 * kt + sr) * D + sc;
      const __bf16* gv = Vt + vbase + (size_t)sr * S + 64 * kt + sc;
      #pragma unroll
      for (int j = 0; j < 4; ++j) {
        async_ld_b128(ldsK + j * 16, gk + j * 8);
        async_ld_b128(ldsV + j * 16, gv + j * 8);
      }
      wait_async0();
    }
    __syncthreads();

    // Scores: B[d][key] = K[key][d] -> contiguous along d in Ks rows
    #pragma unroll
    for (int ns = 0; ns < 4; ++ns) {
      const __bf16* krow = &Ks[ns * 16 + (lane & 15)][(lane < 16) ? 0 : 16];
      v16bf b0 = ld_frag16(krow);
      v16bf b1 = ld_frag16(krow + 32);
      v8f sc2 = {};
      sc2 = wmma_bf16(qa0, b0, sc2);
      sc2 = wmma_bf16(qa1, b1, sc2);

      const int ncol = ns * 16 + (lane & 15);
      const int kj = 64 * kt + ncol;
      #pragma unroll
      for (int r = 0; r < 8; ++r) {
        const int mloc = wv * 16 + r + ((lane < 16) ? 0 : 8);
        const int qi = 64 * qt + mloc;
        Ss[mloc][ncol] = (kj <= qi) ? sc2[r] * 0.125f : -1e30f;
      }
    }
    __syncthreads();

    // Online softmax, one thread per query row; packed bf16 writes to Ps
    if (tid < 64) {
      float mx = mrow[tid];
      for (int j = 0; j < 64; ++j) mx = fmaxf(mx, Ss[tid][j]);
      const float al = __expf(mrow[tid] - mx);
      float sum = 0.f;
      for (int j = 0; j < 64; j += 2) {
        const float p0 = __expf(Ss[tid][j] - mx);
        const float p1 = __expf(Ss[tid][j + 1] - mx);
        *(uint32_t*)&Ps[tid][j] = pk2bf(p0, p1);
        sum += p0 + p1;
      }
      lrow[tid] = lrow[tid] * al + sum;
      mrow[tid] = mx;
      alrow[tid] = al;
    }
    __syncthreads();

    // Rescale running O, then O += P(16x64) x V(64x64)
    const int rbase = wv * 16 + ((lane < 16) ? 0 : 8);
    float4 al0 = *(const float4*)&alrow[rbase];
    float4 al1 = *(const float4*)&alrow[rbase + 4];
    const float alf[8] = { al0.x, al0.y, al0.z, al0.w, al1.x, al1.y, al1.z, al1.w };

    const __bf16* prow = &Ps[wv * 16 + (lane & 15)][0];
    v16bf pa0 = ld_afrag(prow, ab0);
    v16bf pa1 = ld_afrag(prow + 32, ab0);

    #pragma unroll
    for (int ns = 0; ns < 4; ++ns) {
      #pragma unroll
      for (int r = 0; r < 8; ++r) o[ns][r] *= alf[r];
      const __bf16* vrow = &Vs[ns * 16 + (lane & 15)][(lane < 16) ? 0 : 16];
      v16bf b0 = ld_frag16(vrow);
      v16bf b1 = ld_frag16(vrow + 32);
      o[ns] = wmma_bf16(pa0, b0, o[ns]);
      o[ns] = wmma_bf16(pa1, b1, o[ns]);
    }
    __syncthreads();
  }

  // Normalize by l and store
  const int rbase = wv * 16 + ((lane < 16) ? 0 : 8);
  float4 l0 = *(const float4*)&lrow[rbase];
  float4 l1 = *(const float4*)&lrow[rbase + 4];
  const float inv[8] = { 1.f / l0.x, 1.f / l0.y, 1.f / l0.z, 1.f / l0.w,
                         1.f / l1.x, 1.f / l1.y, 1.f / l1.z, 1.f / l1.w };

  #pragma unroll
  for (int ns = 0; ns < 4; ++ns) {
    const int col = ns * 16 + (lane & 15);
    #pragma unroll
    for (int r = 0; r < 8; ++r) {
      const int row = 64 * qt + rbase + r;
      Ob[base + (size_t)row * D + col] = f2bf(o[ns][r] * inv[r]);
    }
  }
}

// ---------------------------------------------------------------------------
// SwiGLU elementwise, 8 elems per thread (uint4 in/out)
// ---------------------------------------------------------------------------
__global__ __launch_bounds__(256)
void swiglu_kernel(const uint4* __restrict__ g, const uint4* __restrict__ u,
                   uint4* __restrict__ out, size_t n8) {
  const size_t i = (size_t)blockIdx.x * blockDim.x + threadIdx.x;
  if (i >= n8) return;
  uint4 gv = g[i], uv = u[i];
  const uint32_t gp[4] = { gv.x, gv.y, gv.z, gv.w };
  const uint32_t up[4] = { uv.x, uv.y, uv.z, uv.w };
  uint32_t op[4];
  #pragma unroll
  for (int j = 0; j < 4; ++j) {
    const float g0 = bflo(gp[j]), g1 = bfhi(gp[j]);
    const float u0 = bflo(up[j]), u1 = bfhi(up[j]);
    const float s0 = g0 / (1.f + __expf(-g0)) * u0;
    const float s1 = g1 / (1.f + __expf(-g1)) * u1;
    op[j] = pk2bf(s0, s1);
  }
  uint4 o; o.x = op[0]; o.y = op[1]; o.z = op[2]; o.w = op[3];
  out[i] = o;
}

// ---------------------------------------------------------------------------
// Host orchestration
// ---------------------------------------------------------------------------
extern "C" void kernel_launch(void* const* d_in, const int* in_sizes, int n_in,
                              void* d_out, int out_size, void* d_ws, size_t ws_size,
                              hipStream_t stream) {
  (void)in_sizes; (void)n_in; (void)out_size; (void)ws_size;
  const float* x   = (const float*)d_in[0];
  const float* W_q = (const float*)d_in[1];
  const float* W_k = (const float*)d_in[2];
  const float* W_v = (const float*)d_in[3];
  const float* W_o = (const float*)d_in[4];
  const float* g1  = (const float*)d_in[5];
  const float* g2  = (const float*)d_in[6];
  const float* W1  = (const float*)d_in[7];
  const float* W2  = (const float*)d_in[8];
  const float* W3  = (const float*)d_in[9];

  constexpr int B = CB, S = CS, D = CD, F = CF, H = CH;
  constexpr int M = B * S;

  char* p = (char*)d_ws;
  auto carve = [&](size_t bytes) -> char* {
    char* r = p;
    p += (bytes + 255) & ~(size_t)255;
    return r;
  };
  __bf16* xb1   = (__bf16*)carve((size_t)M * D * 2);
  __bf16* qb    = (__bf16*)carve((size_t)M * D * 2);
  __bf16* kb    = (__bf16*)carve((size_t)M * D * 2);
  __bf16* vtb   = (__bf16*)carve((size_t)M * D * 2);   // V^T [B][H][64][S]
  __bf16* attnb = (__bf16*)carve((size_t)M * D * 2);
  float*  h1    = (float*) carve((size_t)M * D * 4);
  __bf16* xb2   = (__bf16*)carve((size_t)M * D * 2);
  __bf16* gb    = (__bf16*)carve((size_t)M * F * 2);
  __bf16* ub    = (__bf16*)carve((size_t)M * F * 2);
  __bf16* hb    = (__bf16*)carve((size_t)M * F * 2);

  // 1) pre-attention RMSNorm -> bf16
  rmsnorm_kernel<<<M, 256, 0, stream>>>(x, g1, xb1, D);

  // 2) Q/K/V projections (V stored transposed for the PV matmul)
  dim3 gqkv(D / 128, M / 128);
  gemm_bf16_wmma<1, false><<<gqkv, 256, 0, stream>>>(xb1, W_q, nullptr, qb,  nullptr, M, D, D);
  gemm_bf16_wmma<1, false><<<gqkv, 256, 0, stream>>>(xb1, W_k, nullptr, kb,  nullptr, M, D, D);
  gemm_bf16_wmma<2, false><<<gqkv, 256, 0, stream>>>(xb1, W_v, nullptr, vtb, nullptr, M, D, D);

  // 3) causal flash attention
  attn_kernel<<<dim3(S / 64, H, B), 128, 0, stream>>>(qb, kb, vtb, attnb, S, D);

  // 4) output projection + residual (f32)
  gemm_bf16_wmma<0, true ><<<gqkv, 256, 0, stream>>>(attnb, W_o, h1, nullptr, x, M, D, D);

  // 5) pre-FFN RMSNorm -> bf16
  rmsnorm_kernel<<<M, 256, 0, stream>>>(h1, g2, xb2, D);

  // 6) gate & up projections (bf16 out)
  dim3 gff(F / 128, M / 128);
  gemm_bf16_wmma<1, false><<<gff, 256, 0, stream>>>(xb2, W1, nullptr, gb, nullptr, M, F, D);
  gemm_bf16_wmma<1, false><<<gff, 256, 0, stream>>>(xb2, W3, nullptr, ub, nullptr, M, F, D);

  // 7) SwiGLU combine (8 elems/thread)
  const size_t n8 = (size_t)M * F / 8;
  swiglu_kernel<<<(unsigned)((n8 + 255) / 256), 256, 0, stream>>>(
      (const uint4*)gb, (const uint4*)ub, (uint4*)hb, n8);

  // 8) down projection + residual -> d_out (f32)
  gemm_bf16_wmma<0, true ><<<gqkv, 256, 0, stream>>>(hb, W2, (float*)d_out, nullptr, h1, M, D, F);
}
